// Net_65549790871635
// MI455X (gfx1250) — matrix operand order repaired and checked by reading, not compiled, not run on hardware
//
#include <hip/hip_runtime.h>
#include <hip/hip_bf16.h>
#include <math.h>

// ---------------------------------------------------------------------------
// MI455X (gfx1250) implementation.
// All GEMMs (stem 97->64->32, edge 39(->64)->64->128, big 128->512->512->512->128)
// run on v_wmma_f32_16x16x32_bf16 (wave32, 16x16 tiles, f32 accumulate).
// Weights are swizzled once into per-lane WMMA B-fragment order in d_ws.
// Big-MLP kernel: 32 rows/block, each B fragment reused across 2 row tiles to
// halve L2 weight traffic (arithmetic intensity ~32 FLOP/B vs L2).
// ---------------------------------------------------------------------------

typedef __bf16 bf16;
typedef __attribute__((ext_vector_type(16))) bf16  v16bf;
typedef __attribute__((ext_vector_type(8)))  float v8f;

#define NPTS  131072
#define NBAT  4
#define PERB  32768
#define MROW  32768
#define KNBR  16

#define BNCF 0.9999950000374997f  // 1/sqrt(1+1e-5)

__device__ __forceinline__ float silu_f(float x){ return x / (1.0f + __expf(-x)); }

// ---- WMMA fragment helpers -------------------------------------------------
// A fragment (16x32 bf16) from a row-major LDS tile [16][ld], K-chunk kt.
// ISA layout: lanes 0-15 -> M=lane, VGPR v<4: K=2v,2v+1 ; v>=4: K=16+2(v-4)+{0,1}
//             lanes 16-31 same M, K offset +8. Two contiguous 8-half chunks ->
//             compiler emits 2x ds_load_b128 per fragment.
__device__ __forceinline__ v16bf load_a_frag(const bf16* tile, int ld, int kt, int lane){
  int mrow = lane & 15, hl = lane >> 4;
  const bf16* base = tile + mrow*ld + kt*32;
  v16bf a;
#pragma unroll
  for (int v = 0; v < 8; ++v){
    int k0 = (v < 4) ? (8*hl + 2*v) : (16 + 8*hl + 2*(v-4));
    a[2*v]   = base[k0];
    a[2*v+1] = base[k0+1];
  }
  return a;
}

// B fragment (32x16 bf16) from pre-swizzled weights: lane-contiguous 16 halves
// -> 2x global_load_b128 per fragment.
__device__ __forceinline__ v16bf load_b_frag(const bf16* wsw, int NT, int kt, int nt, int lane){
  const bf16* p = wsw + ((((size_t)kt*NT + nt)*32 + (size_t)lane) << 4);
  v16bf b;
#pragma unroll
  for (int i = 0; i < 16; ++i) b[i] = p[i];
  return b;
}

__device__ __forceinline__ v8f wmma_bf16(v16bf a, v16bf b, v8f c){
  return __builtin_amdgcn_wmma_f32_16x16x32_bf16(false, a, false, b, (short)0, c, false, false);
}

// ---- Kernel W: swizzle f32 row-major weights -> bf16 fragment layout -------
// Fragment layout: half index = (((kt*NT + nt)*32 + lane)*16 + h)
//   K = kt*32 + (h<8 ? 8*(lane>>4)+h : 16+8*(lane>>4)+h-8) ; Nn = nt*16 + (lane&15). Pads with zeros.
__global__ void swz_kernel(const float* __restrict__ W, bf16* __restrict__ dst,
                           int Ktrue, int Ntrue, int KT, int NT){
  int total = KT*NT*512;
  for (int i = blockIdx.x*blockDim.x + threadIdx.x; i < total; i += gridDim.x*blockDim.x){
    int h    = i & 15;
    int lane = (i >> 4) & 31;
    int tn   = i >> 9;
    int nt   = tn % NT, kt = tn / NT;
    int k = kt*32 + ((h < 8) ? (8*(lane>>4) + h) : (16 + 8*(lane>>4) + (h-8)));
    int n = nt*16 + (lane & 15);
    float v = (k < Ktrue && n < Ntrue) ? W[(size_t)k*Ntrue + n] : 0.0f;
    dst[i] = (bf16)v;
  }
}

// ---- Kernel A: per-batch centers of pos/sf (segments are contiguous) -------
__global__ void centers_kernel(const float* __restrict__ pos, const float* __restrict__ sf,
                               float* __restrict__ centers){
  int b = blockIdx.x, tid = threadIdx.x;
  float s0=0.f, s1=0.f, s2=0.f;
  for (int i = tid; i < PERB; i += blockDim.x){
    const float* p = pos + (size_t)(b*PERB + i)*3;
    s0 += p[0]; s1 += p[1]; s2 += p[2];
  }
  __shared__ float red[768];
  red[tid] = s0; red[tid+256] = s1; red[tid+512] = s2;
  __syncthreads();
  for (int off = 128; off > 0; off >>= 1){
    if (tid < off){
      red[tid]     += red[tid+off];
      red[tid+256] += red[tid+256+off];
      red[tid+512] += red[tid+512+off];
    }
    __syncthreads();
  }
  if (tid == 0){
    float inv = 1.0f / (sf[b] * (float)PERB);
    centers[b*3+0] = red[0]  *inv;
    centers[b*3+1] = red[256]*inv;
    centers[b*3+2] = red[512]*inv;
  }
}

// ---- Kernel B: stem (PE + LN97 + MLP 97->64->32 via WMMA) -> x bf16 --------
// 128 threads = 4 waves, each wave owns 16 points.
__global__ void stem_kernel(const float* __restrict__ pos, const float* __restrict__ refl,
                            const float* __restrict__ sf, const int* __restrict__ batch,
                            const float* __restrict__ centers,
                            const float* __restrict__ lng, const float* __restrict__ lnb,
                            const bf16* __restrict__ w1sw, const float* __restrict__ b1,
                            const bf16* __restrict__ w2sw, const float* __restrict__ b2,
                            bf16* __restrict__ xout){
  __shared__ float raw[4][16*97];
  __shared__ bf16  xin[4][16*128];
  __shared__ bf16  hid[4][16*64];
  int tid = threadIdx.x, wave = tid >> 5, lane = tid & 31;
  int j = lane & 15, hl = lane >> 4;
  int i = blockIdx.x*64 + wave*16 + j;
  int b = batch[i];
  float invsf = 1.0f / sf[b];
  float np0 = pos[(size_t)i*3+0]*invsf - centers[b*3+0];
  float np1 = pos[(size_t)i*3+1]*invsf - centers[b*3+1];
  float np2 = pos[(size_t)i*3+2]*invsf - centers[b*3+2];
  float npv[3] = {np0, np1, np2};
  int f0 = hl ? 49 : 0;
  int f1 = hl ? 97 : 49;
  float s1 = 0.f, s2 = 0.f;
  for (int f = f0; f < f1; ++f){
    float v;
    if (f < 96){
      int d = f >> 5, r = f & 31, fi = r & 15;
      float freq = __expf(-(float)fi * 0.125f * 9.210340371976184f); // 10000^(-2fi/16)
      float arg = npv[d] * freq;
      v = (r < 16) ? __sinf(arg) : __cosf(arg);
    } else {
      v = refl[i];
    }
    raw[wave][j*97 + f] = v;
    s1 += v; s2 += v*v;
  }
  s1 += __shfl_xor(s1, 16);
  s2 += __shfl_xor(s2, 16);
  float mean = s1 * (1.0f/97.0f);
  float var  = s2 * (1.0f/97.0f) - mean*mean;
  float rstd = rsqrtf(var + 1e-5f);
  for (int f = f0; f < f1; ++f){
    float v = (raw[wave][j*97 + f] - mean)*rstd*lng[f] + lnb[f];
    xin[wave][j*128 + f] = (bf16)v;
  }
  if (hl){ for (int f = 97; f < 128; ++f) xin[wave][j*128 + f] = (bf16)0.0f; }
  __syncthreads();

  // GEMM1: (16x128) @ (128x64), K padded 97->128 with zero rows.
  v8f acc[4];
#pragma unroll
  for (int t = 0; t < 4; ++t){ v8f z = {}; acc[t] = z; }
  for (int kt = 0; kt < 4; ++kt){
    v16bf a = load_a_frag(&xin[wave][0], 128, kt, lane);
#pragma unroll
    for (int nt = 0; nt < 4; ++nt)
      acc[nt] = wmma_bf16(a, load_b_frag(w1sw, 4, kt, nt, lane), acc[nt]);
  }
#pragma unroll
  for (int nt = 0; nt < 4; ++nt){
    int n = nt*16 + (lane & 15);
#pragma unroll
    for (int r = 0; r < 8; ++r){
      float o = silu_f(BNCF*(acc[nt][r] + b1[n]));
      hid[wave][(r + 8*hl)*64 + n] = (bf16)o;
    }
  }
  __syncthreads();

  // GEMM2: (16x64) @ (64x32)
  v8f acc2[2];
#pragma unroll
  for (int t = 0; t < 2; ++t){ v8f z = {}; acc2[t] = z; }
  for (int kt = 0; kt < 2; ++kt){
    v16bf a = load_a_frag(&hid[wave][0], 64, kt, lane);
#pragma unroll
    for (int nt = 0; nt < 2; ++nt)
      acc2[nt] = wmma_bf16(a, load_b_frag(w2sw, 2, kt, nt, lane), acc2[nt]);
  }
#pragma unroll
  for (int nt = 0; nt < 2; ++nt){
    int n = nt*16 + (lane & 15);
#pragma unroll
    for (int r = 0; r < 8; ++r){
      int m = blockIdx.x*64 + wave*16 + r + 8*hl;
      float o = silu_f(BNCF*(acc2[nt][r] + b2[n]));
      xout[(size_t)m*32 + n] = (bf16)o;
    }
  }
}

// ---- Kernel C: per-row neighbor direction stats -> sfeat (M x 3) -----------
__device__ __constant__ float KD[24] = {
   0.5773502692f,  0.5773502692f,  0.5773502692f,
  -0.5773502692f,  0.5773502692f,  0.5773502692f,
   0.5773502692f, -0.5773502692f,  0.5773502692f,
   0.5773502692f,  0.5773502692f, -0.5773502692f,
  -0.5773502692f, -0.5773502692f,  0.5773502692f,
  -0.5773502692f,  0.5773502692f, -0.5773502692f,
   0.5773502692f, -0.5773502692f, -0.5773502692f,
  -0.5773502692f, -0.5773502692f, -0.5773502692f };

__global__ void sfeat_kernel(const float* __restrict__ pos,
                             const int* __restrict__ idx, const int* __restrict__ col,
                             const float* __restrict__ lkw, const float* __restrict__ lkb,
                             const float* __restrict__ lng, const float* __restrict__ lnb,
                             float* __restrict__ sfeat){
  int m = blockIdx.x*blockDim.x + threadIdx.x;
  if (m >= MROW) return;
  int im = idx[m];
  float px = pos[(size_t)im*3+0], py = pos[(size_t)im*3+1], pz = pos[(size_t)im*3+2];
  float nb[8] = {0,0,0,0,0,0,0,0};
  for (int j = 0; j < KNBR; ++j){
    int c = col[(size_t)m*KNBR + j];
    float dx = pos[(size_t)c*3+0] - px;
    float dy = pos[(size_t)c*3+1] - py;
    float dz = pos[(size_t)c*3+2] - pz;
    float inv = 1.0f / (sqrtf(dx*dx + dy*dy + dz*dz) + 1e-8f);
    float ux = dx*inv, uy = dy*inv, uz = dz*inv;
#pragma unroll
    for (int q = 0; q < 8; ++q)
      nb[q] += KD[q*3+0]*ux + KD[q*3+1]*uy + KD[q*3+2]*uz;
  }
  float s[3];
#pragma unroll
  for (int c3 = 0; c3 < 3; ++c3){
    float t = lkb[c3];
#pragma unroll
    for (int q = 0; q < 8; ++q) t += (nb[q]*(1.0f/16.0f)) * lkw[q*3 + c3];
    s[c3] = t;
  }
  float mean = (s[0]+s[1]+s[2]) * (1.0f/3.0f);
  float var  = ((s[0]-mean)*(s[0]-mean) + (s[1]-mean)*(s[1]-mean) + (s[2]-mean)*(s[2]-mean)) * (1.0f/3.0f);
  float rstd = rsqrtf(var + 1e-5f);
#pragma unroll
  for (int c3 = 0; c3 < 3; ++c3)
    sfeat[(size_t)m*3 + c3] = (s[c3]-mean)*rstd*lng[c3] + lnb[c3];
}

// ---- Kernel D: edge MLP 64(padded 39)->64->128 + per-row max -> agg --------
// 256 threads = 8 waves; each wave owns one output row (its 16 contiguous edges).
__global__ void edge_kernel(const float* __restrict__ pos, const float* __restrict__ refl,
                            const int* __restrict__ idx, const int* __restrict__ col,
                            const bf16* __restrict__ x, const float* __restrict__ sfeat,
                            const bf16* __restrict__ w1sw, const float* __restrict__ b1,
                            const bf16* __restrict__ w2sw, const float* __restrict__ b2,
                            float* __restrict__ agg){
  __shared__ bf16 msg[8][16*64];
  __shared__ bf16 hid[8][16*64];
  int tid = threadIdx.x, wave = tid >> 5, lane = tid & 31;
  int m = blockIdx.x*8 + wave;
  int j = lane >> 1, fh = lane & 1;
  int e = m*KNBR + j;
  int c = col[e];
  if (fh == 0){
    const bf16* xr = x + (size_t)c*32;
#pragma unroll
    for (int f = 0; f < 32; ++f) msg[wave][j*64 + f] = xr[f];
  } else {
    int im = idx[m];
    float r0 = pos[(size_t)c*3+0] - pos[(size_t)im*3+0];
    float r1 = pos[(size_t)c*3+1] - pos[(size_t)im*3+1];
    float r2 = pos[(size_t)c*3+2] - pos[(size_t)im*3+2];
    float r3 = refl[c] - refl[im];
    msg[wave][j*64+32] = (bf16)r0;
    msg[wave][j*64+33] = (bf16)r1;
    msg[wave][j*64+34] = (bf16)r2;
    msg[wave][j*64+35] = (bf16)r3;
    msg[wave][j*64+36] = (bf16)sfeat[(size_t)m*3+0];
    msg[wave][j*64+37] = (bf16)sfeat[(size_t)m*3+1];
    msg[wave][j*64+38] = (bf16)sfeat[(size_t)m*3+2];
    for (int f = 39; f < 64; ++f) msg[wave][j*64 + f] = (bf16)0.0f;
  }
  __syncthreads();

  // Layer 1: (16x64) @ (64x64)  (K padded 39->64 with zero weight rows)
  v8f acc1[4];
#pragma unroll
  for (int t = 0; t < 4; ++t){ v8f z = {}; acc1[t] = z; }
  for (int kt = 0; kt < 2; ++kt){
    v16bf a = load_a_frag(&msg[wave][0], 64, kt, lane);
#pragma unroll
    for (int nt = 0; nt < 4; ++nt)
      acc1[nt] = wmma_bf16(a, load_b_frag(w1sw, 4, kt, nt, lane), acc1[nt]);
  }
  int hl = lane >> 4;
#pragma unroll
  for (int nt = 0; nt < 4; ++nt){
    int n = nt*16 + (lane & 15);
#pragma unroll
    for (int r = 0; r < 8; ++r){
      float o = silu_f(BNCF*(acc1[nt][r] + b1[n]));
      hid[wave][(r + 8*hl)*64 + n] = (bf16)o;
    }
  }
  __syncthreads();

  // Layer 2: (16x64) @ (64x128), then max over the 16 edges -> agg row
  v8f acc2[8];
#pragma unroll
  for (int t = 0; t < 8; ++t){ v8f z = {}; acc2[t] = z; }
  for (int kt = 0; kt < 2; ++kt){
    v16bf a = load_a_frag(&hid[wave][0], 64, kt, lane);
#pragma unroll
    for (int nt = 0; nt < 8; ++nt)
      acc2[nt] = wmma_bf16(a, load_b_frag(w2sw, 8, kt, nt, lane), acc2[nt]);
  }
#pragma unroll
  for (int nt = 0; nt < 8; ++nt){
    int n = nt*16 + (lane & 15);
    float mx = -1e30f;
#pragma unroll
    for (int r = 0; r < 8; ++r)
      mx = fmaxf(mx, silu_f(BNCF*(acc2[nt][r] + b2[n])));
    mx = fmaxf(mx, __shfl_xor(mx, 16));
    if (lane < 16) agg[(size_t)m*128 + n] = mx;
  }
}

// ---- Kernel E: fused big MLP 128->512->512->512->128 + residual silu -------
// 256 threads = 8 waves; block owns 32 rows (2 row tiles); each wave owns 64
// output columns for the 512-wide layers. Each B fragment is loaded once and
// used by 2 WMMAs (two row tiles) -> halves L2 weight traffic.
// LDS: two 32x512 bf16 buffers (64KB); the 32x128 input tile aliases actG,
// which is first written only after the input tile is dead.
__global__ void mlp_kernel(const float* __restrict__ agg,
                           const bf16* __restrict__ expsw, const float* __restrict__ expb,
                           const float* __restrict__ dw1w, const float* __restrict__ dw1b,
                           const bf16* __restrict__ pw1sw, const float* __restrict__ pw1b,
                           const float* __restrict__ dw2w, const float* __restrict__ dw2b,
                           const bf16* __restrict__ pw2sw, const float* __restrict__ pw2b,
                           const bf16* __restrict__ projsw, const float* __restrict__ projb,
                           float* __restrict__ out){
  __shared__ bf16 actH[32*512];
  __shared__ bf16 actG[32*512];
  bf16* actA = actG;   // 32x128 input tile alias (dead before actG is written)
  int tid = threadIdx.x, wave = tid >> 5, lane = tid & 31;
  int hl = lane >> 4;
  int m0 = blockIdx.x*32;
  for (int i = tid; i < 32*128; i += 256)
    actA[i] = (bf16)agg[(size_t)m0*128 + i];
  __syncthreads();

  v8f acc[2][4];
  // Layer exp: (32x128)@(128x512)
#pragma unroll
  for (int rt = 0; rt < 2; ++rt)
#pragma unroll
    for (int t = 0; t < 4; ++t){ v8f z = {}; acc[rt][t] = z; }
  for (int kt = 0; kt < 4; ++kt){
    v16bf a0 = load_a_frag(actA,        128, kt, lane);
    v16bf a1 = load_a_frag(actA + 2048, 128, kt, lane);
#pragma unroll
    for (int t = 0; t < 4; ++t){
      v16bf bfr = load_b_frag(expsw, 32, kt, wave*4 + t, lane);
      acc[0][t] = wmma_bf16(a0, bfr, acc[0][t]);
      acc[1][t] = wmma_bf16(a1, bfr, acc[1][t]);
    }
  }
#pragma unroll
  for (int rt = 0; rt < 2; ++rt)
#pragma unroll
    for (int t = 0; t < 4; ++t){
      int n = (wave*4 + t)*16 + (lane & 15);
#pragma unroll
      for (int r = 0; r < 8; ++r){
        float o = silu_f(BNCF*(acc[rt][t][r] + expb[n]));
        o = silu_f(BNCF*(o*dw1w[n] + dw1b[n]));
        actH[(rt*16 + r + 8*hl)*512 + n] = (bf16)o;
      }
    }
  __syncthreads();

  // Layer pw1: (32x512)@(512x512)
#pragma unroll
  for (int rt = 0; rt < 2; ++rt)
#pragma unroll
    for (int t = 0; t < 4; ++t){ v8f z = {}; acc[rt][t] = z; }
  for (int kt = 0; kt < 16; ++kt){
    v16bf a0 = load_a_frag(actH,        512, kt, lane);
    v16bf a1 = load_a_frag(actH + 8192, 512, kt, lane);
#pragma unroll
    for (int t = 0; t < 4; ++t){
      v16bf bfr = load_b_frag(pw1sw, 32, kt, wave*4 + t, lane);
      acc[0][t] = wmma_bf16(a0, bfr, acc[0][t]);
      acc[1][t] = wmma_bf16(a1, bfr, acc[1][t]);
    }
  }
#pragma unroll
  for (int rt = 0; rt < 2; ++rt)
#pragma unroll
    for (int t = 0; t < 4; ++t){
      int n = (wave*4 + t)*16 + (lane & 15);
#pragma unroll
      for (int r = 0; r < 8; ++r){
        float o = silu_f(BNCF*(acc[rt][t][r] + pw1b[n]));
        o = silu_f(BNCF*o);
        o = silu_f(BNCF*(o*dw2w[n] + dw2b[n]));
        actG[(rt*16 + r + 8*hl)*512 + n] = (bf16)o;
      }
    }
  __syncthreads();

  // Layer pw2: (32x512)@(512x512)
#pragma unroll
  for (int rt = 0; rt < 2; ++rt)
#pragma unroll
    for (int t = 0; t < 4; ++t){ v8f z = {}; acc[rt][t] = z; }
  for (int kt = 0; kt < 16; ++kt){
    v16bf a0 = load_a_frag(actG,        512, kt, lane);
    v16bf a1 = load_a_frag(actG + 8192, 512, kt, lane);
#pragma unroll
    for (int t = 0; t < 4; ++t){
      v16bf bfr = load_b_frag(pw2sw, 32, kt, wave*4 + t, lane);
      acc[0][t] = wmma_bf16(a0, bfr, acc[0][t]);
      acc[1][t] = wmma_bf16(a1, bfr, acc[1][t]);
    }
  }
#pragma unroll
  for (int rt = 0; rt < 2; ++rt)
#pragma unroll
    for (int t = 0; t < 4; ++t){
      int n = (wave*4 + t)*16 + (lane & 15);
#pragma unroll
      for (int r = 0; r < 8; ++r){
        float o = silu_f(BNCF*(acc[rt][t][r] + pw2b[n]));
        o = BNCF*o;
        actH[(rt*16 + r + 8*hl)*512 + n] = (bf16)o;
      }
    }
  __syncthreads();

  // Layer proj: (32x512)@(512x128) + residual + silu. Each wave owns 1 N-tile.
  v8f accp[2];
#pragma unroll
  for (int rt = 0; rt < 2; ++rt){ v8f z = {}; accp[rt] = z; }
  for (int kt = 0; kt < 16; ++kt){
    v16bf a0 = load_a_frag(actH,        512, kt, lane);
    v16bf a1 = load_a_frag(actH + 8192, 512, kt, lane);
    v16bf bfr = load_b_frag(projsw, 8, kt, wave, lane);
    accp[0] = wmma_bf16(a0, bfr, accp[0]);
    accp[1] = wmma_bf16(a1, bfr, accp[1]);
  }
#pragma unroll
  for (int rt = 0; rt < 2; ++rt){
    int n = wave*16 + (lane & 15);
#pragma unroll
    for (int r = 0; r < 8; ++r){
      int m = m0 + rt*16 + r + 8*hl;
      float o = BNCF*(accp[rt][r] + projb[n]);
      float res = agg[(size_t)m*128 + n];
      out[(size_t)m*128 + n] = silu_f(o + res);
    }
  }
}

// ---------------------------------------------------------------------------
extern "C" void kernel_launch(void* const* d_in, const int* in_sizes, int n_in,
                              void* d_out, int out_size, void* d_ws, size_t ws_size,
                              hipStream_t stream) {
  (void)in_sizes; (void)n_in; (void)out_size; (void)ws_size;
  const float* pos   = (const float*)d_in[0];
  const float* refl  = (const float*)d_in[1];
  const float* sf    = (const float*)d_in[2];
  const int*   batch = (const int*)d_in[3];
  const int*   idx   = (const int*)d_in[4];
  const int*   col   = (const int*)d_in[5];
  // d_in[6] = row (implicit: edges of row m are m*16..m*16+15)
  const float* stem_ln_g = (const float*)d_in[7];
  const float* stem_ln_b = (const float*)d_in[8];
  const float* stem_w1   = (const float*)d_in[9];
  const float* stem_b1   = (const float*)d_in[10];
  const float* stem_w2   = (const float*)d_in[11];
  const float* stem_b2   = (const float*)d_in[12];
  const float* lk_w      = (const float*)d_in[13];
  const float* lk_b      = (const float*)d_in[14];
  const float* lk_ln_g   = (const float*)d_in[15];
  const float* lk_ln_b   = (const float*)d_in[16];
  const float* loc_w1    = (const float*)d_in[17];
  const float* loc_b1    = (const float*)d_in[18];
  const float* loc_w2    = (const float*)d_in[19];
  const float* loc_b2    = (const float*)d_in[20];
  const float* exp_w     = (const float*)d_in[21];
  const float* exp_b     = (const float*)d_in[22];
  const float* dw1_w     = (const float*)d_in[23];
  const float* dw1_b     = (const float*)d_in[24];
  const float* pw1_w     = (const float*)d_in[25];
  const float* pw1_b     = (const float*)d_in[26];
  const float* dw2_w     = (const float*)d_in[27];
  const float* dw2_b     = (const float*)d_in[28];
  const float* pw2_w     = (const float*)d_in[29];
  const float* pw2_b     = (const float*)d_in[30];
  const float* proj_w    = (const float*)d_in[31];
  const float* proj_b    = (const float*)d_in[32];
  float* out = (float*)d_out;

  // ---- workspace layout ----
  char* ws = (char*)d_ws;
  float* centers = (float*)(ws);                                   // 48 B (pad 256)
  size_t off = 256;
  bf16*  xbf   = (bf16*)(ws + off);  off += (size_t)NPTS*32*2;     // 8.39 MB
  float* sfeat = (float*)(ws + off); off += (size_t)MROW*3*4;      // 0.39 MB
  float* agg   = (float*)(ws + off); off += (size_t)MROW*128*4;    // 16.8 MB
  bf16* stemw1sw = (bf16*)(ws + off);                              // 4*4*512 halves
  bf16* stemw2sw = stemw1sw + 8192;                                // 2*2*512
  bf16* locw1sw  = stemw2sw + 2048;                                // 2*4*512
  bf16* locw2sw  = locw1sw  + 4096;                                // 2*8*512
  bf16* expsw    = locw2sw  + 8192;                                // 4*32*512
  bf16* pw1sw    = expsw    + 65536;                               // 16*32*512
  bf16* pw2sw    = pw1sw    + 262144;
  bf16* projsw   = pw2sw    + 262144;                              // 16*8*512

  // ---- one-time (per launch) weight swizzles into WMMA fragment order ----
  swz_kernel<<<32,   256, 0, stream>>>(stem_w1, stemw1sw,  97,  64,  4,  4);
  swz_kernel<<<8,    256, 0, stream>>>(stem_w2, stemw2sw,  64,  32,  2,  2);
  swz_kernel<<<16,   256, 0, stream>>>(loc_w1,  locw1sw,   39,  64,  2,  4);
  swz_kernel<<<32,   256, 0, stream>>>(loc_w2,  locw2sw,   64, 128,  2,  8);
  swz_kernel<<<256,  256, 0, stream>>>(exp_w,   expsw,    128, 512,  4, 32);
  swz_kernel<<<1024, 256, 0, stream>>>(pw1_w,   pw1sw,    512, 512, 16, 32);
  swz_kernel<<<1024, 256, 0, stream>>>(pw2_w,   pw2sw,    512, 512, 16, 32);
  swz_kernel<<<256,  256, 0, stream>>>(proj_w,  projsw,   512, 128, 16,  8);

  centers_kernel<<<NBAT, 256, 0, stream>>>(pos, sf, centers);

  stem_kernel<<<NPTS/64, 128, 0, stream>>>(pos, refl, sf, batch, centers,
                                           stem_ln_g, stem_ln_b,
                                           stemw1sw, stem_b1, stemw2sw, stem_b2, xbf);

  sfeat_kernel<<<MROW/256, 256, 0, stream>>>(pos, idx, col, lk_w, lk_b,
                                             lk_ln_g, lk_ln_b, sfeat);

  edge_kernel<<<MROW/8, 256, 0, stream>>>(pos, refl, idx, col, xbf, sfeat,
                                          locw1sw, loc_b1, locw2sw, loc_b2, agg);

  mlp_kernel<<<MROW/32, 256, 0, stream>>>(agg,
                                          expsw, exp_b, dw1_w, dw1_b,
                                          pw1sw, pw1_b, dw2_w, dw2_b,
                                          pw2sw, pw2_b, projsw, proj_b, out);
}
